// QuantumBoltzmannMachine_18356690223150
// MI455X (gfx1250) — matrix operand order repaired and checked by reading, not compiled, not run on hardware
//
#include <hip/hip_runtime.h>
#include <hip/hip_bf16.h>
#include <math.h>

// ---------------------------------------------------------------------------
// QBM fused kernel for MI455X (gfx1250, wave32, WMMA + TDM).
//   enc = tanh(x[32768,2048] @ We[2048,256] + be)   -> bf16 WMMA 16x16x32, f32 acc
//   head: p = enc@W1e; h = tanh(tanh(p + W1i + b1)); en = 3*tanh(h.w2 + b2)
//   MCMC: 100-step Metropolis over E=16 states, threefry2x32 PRNG
// Outputs (concat): probabilities (B,16) | energies (B,16) | logits (B,16)
// Block = 128 threads (4 waves); wave owns 2 M-tiles x 8 N-tiles (16 WMMA/kstep,
// B fragments reused 2x). sched_group_barrier pins loads-then-WMMA pipeline.
// ---------------------------------------------------------------------------

typedef __attribute__((ext_vector_type(16))) __bf16 v16bf;
typedef __attribute__((ext_vector_type(8)))  float  v8f;
typedef __attribute__((ext_vector_type(4)))  unsigned v4u;
typedef __attribute__((ext_vector_type(8)))  int      v8i_;
typedef __attribute__((ext_vector_type(4)))  int      v4i_;

#define NB 32768
#define VD 2048
#define HD 256
#define ED 16
#define NSTEP 100

#define KCHUNK 64
#define BT_PITCH 72   // bf16 elems per B row (64 + 8 pad = 144B, kills bank conflicts)
#define AT_PITCH 72

#define BT_OFF   0
#define BT_BYTES (256 * BT_PITCH * 2)        // 36864
#define AT_OFF   BT_BYTES
#define AT_BYTES (64 * AT_PITCH * 2)         // 9216
#define PH_OFF   (AT_OFF + AT_BYTES)         // 46080: partial p, [2][64][4] f32
#define PH_BYTES (2 * 64 * 4 * 4)
#define KEY_OFF  (PH_OFF + PH_BYTES)         // 48128: per-step MCMC keys
#define KEY_BYTES (NSTEP * 4 * 4)
#define TV_OFF   (KEY_OFF + KEY_BYTES)       // 49728: per-step 1/T
#define SMEM_BYTES 50176                     // < 64KB, no opt-in needed

#if __has_builtin(__builtin_amdgcn_tensor_load_to_lds) && \
    __has_builtin(__builtin_amdgcn_s_wait_tensorcnt)
#define USE_TDM 1
#else
#define USE_TDM 0
#endif

#if __has_builtin(__builtin_amdgcn_sched_group_barrier)
#define SGB(mask, cnt) __builtin_amdgcn_sched_group_barrier((mask), (cnt), 0)
#else
#define SGB(mask, cnt)
#endif

__device__ __forceinline__ unsigned pk_bf16(float lo, float hi) {
  unsigned a = __builtin_bit_cast(unsigned, lo);
  unsigned b = __builtin_bit_cast(unsigned, hi);
  a = (a + 0x7FFFu + ((a >> 16) & 1u)) >> 16;   // RNE f32 -> bf16
  b = (b + 0x7FFFu + ((b >> 16) & 1u)) >> 16;
  return a | (b << 16);
}

// threefry2x32, 20 rounds (JAX's PRNG core)
__device__ __forceinline__ uint2 tf2x32(uint2 k, uint2 c) {
  unsigned ks0 = k.x, ks1 = k.y, ks2 = ks0 ^ ks1 ^ 0x1BD11BDAu;
  unsigned x0 = c.x + ks0, x1 = c.y + ks1;
#define TF_R4(a,b,cc,d)                                   \
  x0 += x1; x1 = (x1 << (a)) | (x1 >> (32-(a))); x1 ^= x0; \
  x0 += x1; x1 = (x1 << (b)) | (x1 >> (32-(b))); x1 ^= x0; \
  x0 += x1; x1 = (x1 << (cc))| (x1 >> (32-(cc))); x1 ^= x0; \
  x0 += x1; x1 = (x1 << (d)) | (x1 >> (32-(d))); x1 ^= x0;
  TF_R4(13,15,26,6)  x0 += ks1; x1 += ks2 + 1u;
  TF_R4(17,29,16,24) x0 += ks2; x1 += ks0 + 2u;
  TF_R4(13,15,26,6)  x0 += ks0; x1 += ks1 + 3u;
  TF_R4(17,29,16,24) x0 += ks1; x1 += ks2 + 4u;
  TF_R4(13,15,26,6)  x0 += ks2; x1 += ks0 + 5u;
#undef TF_R4
  return make_uint2(x0, x1);
}

// JAX random_bits(key, 32, (B,)) element b: counters split in halves/pairs
__device__ __forceinline__ unsigned row_bits(uint2 key, unsigned b) {
  const unsigned HB = NB / 2;
  if (b < HB) return tf2x32(key, make_uint2(b, b + HB)).x;
  return tf2x32(key, make_uint2(b - HB, b)).y;
}

// ---------------------------------------------------------------------------
// Kernel 0: WeT[n][k] = bf16(We[k][n])  (1 MB in workspace; lives in L2)
// ---------------------------------------------------------------------------
__global__ void qbm_prep_weT(const float* __restrict__ We,
                             unsigned* __restrict__ WeT) {
  unsigned idx = blockIdx.x * blockDim.x + threadIdx.x;   // one u32 (2 bf16)
  if (idx >= (unsigned)(HD * VD / 2)) return;
  unsigned n  = idx / (VD / 2);
  unsigned k2 = (idx % (VD / 2)) * 2;
  float a = We[(size_t)k2 * HD + n];
  float b = We[(size_t)(k2 + 1) * HD + n];
  WeT[idx] = pk_bf16(a, b);
}

// ---------------------------------------------------------------------------
// Kernel 1: fused GEMM (bf16 WMMA) + head + MCMC sampling
// 512 blocks x 128 threads; tile 64 rows x 256 cols, K chunked by 64.
// wave w: mpair = w&1 (mtiles 2*mpair, 2*mpair+1), nhalf = w>>1 (8 N-tiles).
// B tile staged by the Tensor Data Mover (wave 0), A tile converted by VALU.
// ---------------------------------------------------------------------------
__global__ void __launch_bounds__(128)
qbm_main(const float* __restrict__ x,
         const float* __restrict__ be,
         const float* __restrict__ W1,
         const float* __restrict__ b1,
         const float* __restrict__ w2,
         const float* __restrict__ b2p,
         const float* __restrict__ inv_temp,
         const __bf16* __restrict__ WeT,
         float* __restrict__ out) {
  extern __shared__ char smem[];
  __bf16*   BT   = (__bf16*)(smem + BT_OFF);
  __bf16*   AT   = (__bf16*)(smem + AT_OFF);
  float*    PH   = (float*)(smem + PH_OFF);
  unsigned* KEYS = (unsigned*)(smem + KEY_OFF);
  float*    TINV = (float*)(smem + TV_OFF);

  const int tid   = threadIdx.x;
  const int lane  = tid & 31;
  const int wv    = tid >> 5;
  const int lr    = lane & 15;   // fragment row/col within lane group
  const int lh    = lane >> 4;   // which K half this lane holds
  const int mpair = wv & 1;      // mtiles 2*mpair, 2*mpair+1
  const int nhalf = wv >> 1;     // N-tiles nhalf*8 .. +8
  const int m0    = blockIdx.x * 64;

  v8f acc[2][8] = {};

  // cooperative staging mapping (A tile): 128 threads, 64 rows x 64 k f32
  const int arow = tid >> 1;     // 0..63: x row within tile
  const int aseg = tid & 1;      // 32-f32 segment within 64-wide K chunk
  const float* xsrc = x + (size_t)(m0 + arow) * VD + aseg * 32;

#if USE_TDM
  // D# group1 (constant): data_size=2B, pad 4 dwords every 32 dwords
  // (reproduces BT_PITCH=72), tensor 2048x256, tile 64x256, stride 2048.
  const v8i_ g1 = {
      (int)((1u << 16) | (1u << 20) | (4u << 22) | (3u << 25)), // size/pad ctl
      (int)(2048u << 16),            // tensor_dim0[15:0] in [31:16]
      (int)(256u << 16),             // tensor_dim0 hi=0, tensor_dim1=256
      (int)(64u << 16),              // tensor_dim1 hi=0, tile_dim0=64
      (int)256,                      // tile_dim1=256, tile_dim2=0
      (int)2048,                     // tensor_dim0_stride=2048
      0, 0                           // stride hi, tensor_dim1_stride=0
  };
  const v4i_ z4 = {0, 0, 0, 0};
#if defined(__clang_major__) && (__clang_major__ >= 23)
  const v8i_ z8 = {0, 0, 0, 0, 0, 0, 0, 0};
#endif
#endif

  for (int kc = 0; kc < VD; kc += KCHUNK) {
    __syncthreads();   // previous chunk's LDS reads done

#if USE_TDM
    if (wv == 0) {     // wave-uniform branch: TDM ignores EXEC, issue once
      unsigned long long ga =
          (unsigned long long)(const void*)(WeT + kc);     // tile start (bytes)
      v4u g0 = { 1u,                                       // count=1 valid D#
                 (unsigned)BT_OFF,                         // lds_addr
                 (unsigned)(ga & 0xFFFFFFFFull),           // global_addr[31:0]
                 (unsigned)((ga >> 32) & 0x1FFFFFFull) | (2u << 30) }; // +type=2
#if defined(__clang_major__) && (__clang_major__ >= 23)
      __builtin_amdgcn_tensor_load_to_lds(g0, g1, z4, z4, z8, 0);
#else
      __builtin_amdgcn_tensor_load_to_lds(g0, g1, z4, z4, 0);
#endif
    }
#else
    // stage B tile manually: BT[n][0..64) = WeT[n][kc..kc+64), 2 rows/thread
    {
#pragma unroll
      for (int rr = 0; rr < 256; rr += 128) {
        const float4* s = (const float4*)(WeT + (size_t)(tid + rr) * VD + kc);
        float4* d = (float4*)(BT + (size_t)(tid + rr) * BT_PITCH);
#pragma unroll
        for (int i = 0; i < 8; ++i) d[i] = s[i];
      }
    }
#endif

    // stage A tile: AT[r][0..64) = bf16(x[m0+r][kc..kc+64))
    {
      const float4* s = (const float4*)(xsrc + kc);
      unsigned* d = (unsigned*)(AT + (size_t)arow * AT_PITCH + aseg * 32);
#pragma unroll
      for (int i = 0; i < 4; ++i) {
        float4 fa = s[2 * i], fb = s[2 * i + 1];
        d[4 * i + 0] = pk_bf16(fa.x, fa.y); d[4 * i + 1] = pk_bf16(fa.z, fa.w);
        d[4 * i + 2] = pk_bf16(fb.x, fb.y); d[4 * i + 3] = pk_bf16(fb.z, fb.w);
      }
    }
    if (kc + KCHUNK < VD)
      __builtin_prefetch(xsrc + kc + KCHUNK, 0, 3);  // global_prefetch next chunk

#if USE_TDM
    if (wv == 0) __builtin_amdgcn_s_wait_tensorcnt(0);  // TDM done before barrier
#endif
    __syncthreads();

#pragma unroll
    for (int ks = 0; ks < KCHUNK; ks += 32) {
      union F { v16bf v; float4 f[2]; };
      // A fragments (16x32 bf16): lane lr = row, two contiguous 8-elem K runs
      F af[2];
#pragma unroll
      for (int i = 0; i < 2; ++i) {
        const char* ap = (const char*)(AT
            + (size_t)((mpair * 2 + i) * 16 + lr) * AT_PITCH + ks + lh * 8);
        af[i].f[0] = *(const float4*)ap;          // K = ks + lh*8 .. +8
        af[i].f[1] = *(const float4*)(ap + 32);   // K = ks + 16 + lh*8 .. +8
      }
      // All 8 B fragments (32x16 bf16): lane lr = col N, two 8-elem K runs (same as A)
      F bf[8];
#pragma unroll
      for (int j = 0; j < 8; ++j) {
        const char* bp = (const char*)(BT
            + (size_t)((nhalf * 8 + j) * 16 + lr) * BT_PITCH + ks + lh * 8);
        bf[j].f[0] = *(const float4*)bp;
        bf[j].f[1] = *(const float4*)(bp + 32);
      }
#pragma unroll
      for (int j = 0; j < 8; ++j)
#pragma unroll
        for (int i = 0; i < 2; ++i)
          acc[i][j] = __builtin_amdgcn_wmma_f32_16x16x32_bf16(
              false, af[i].v, false, bf[j].v, (short)0, acc[i][j], false, false);
      // Pin pipeline: 20 DS reads first, then 16 WMMAs. In-order DS returns
      // let the waitcnt pass use partial s_wait_dscnt and overlap WMMA with
      // outstanding LDS traffic (scheduler otherwise sinks loads to uses).
      SGB(0x100, 20);   // DS read group
      SGB(0x008, 16);   // MFMA/WMMA group
    }
  }

  // ---- epilogue: enc = tanh(acc + be); partial p = enc @ W1e (in-register) --
  float pq[2][8][4] = {};
#pragma unroll
  for (int j = 0; j < 8; ++j) {
    int n = (nhalf * 8 + j) * 16 + lr;       // C/D layout: lane lr = column n
    float4 w1e = *(const float4*)(W1 + n * 4);
    float bej = be[n];
#pragma unroll
    for (int i = 0; i < 2; ++i)
#pragma unroll
      for (int r = 0; r < 8; ++r) {          // vgpr r = row (2*mpair+i)*16+r(+8)
        float v = tanhf(acc[i][j][r] + bej);
        pq[i][r][0] += v * w1e.x; pq[i][r][1] += v * w1e.y;
        pq[i][r][2] += v * w1e.z; pq[i][r][3] += v * w1e.w;
      }
  }
  // reduce over the 16 N-lanes (rows for lanes 0-15 vs 16-31 stay separate)
#pragma unroll
  for (int msk = 1; msk <= 8; msk <<= 1)
#pragma unroll
    for (int i = 0; i < 2; ++i)
#pragma unroll
      for (int r = 0; r < 8; ++r)
#pragma unroll
        for (int q = 0; q < 4; ++q)
          pq[i][r][q] += __shfl_xor(pq[i][r][q], msk, 32);
  if (lr == 0) {
#pragma unroll
    for (int i = 0; i < 2; ++i) {
      int mb = (mpair * 2 + i) * 16 + lh * 8;
#pragma unroll
      for (int r = 0; r < 8; ++r)
#pragma unroll
        for (int q = 0; q < 4; ++q)
          PH[((nhalf * 64) + mb + r) * 4 + q] = pq[i][r][q];
    }
  }

  // ---- per-step MCMC keys (threads 0..99), emulating JAX key derivation ----
  if (tid < NSTEP) {
    uint2 root = make_uint2(0u, 1u);                       // jax.random.key(1)
    uint2 o02 = tf2x32(root, make_uint2(0u, 2u));
    uint2 o13 = tf2x32(root, make_uint2(1u, 3u));
    uint2 kloop = make_uint2(o02.y, o13.y);                // split -> (kinit,kloop)
    uint2 kt  = tf2x32(kloop, make_uint2(0u, (unsigned)tid));   // fold_in(kloop,t)
    uint2 p02 = tf2x32(kt, make_uint2(0u, 2u));
    uint2 p13 = tf2x32(kt, make_uint2(1u, 3u));
    uint2 kp  = make_uint2(p02.x, p13.x);                  // split(kt) -> kp,ku
    uint2 ku  = make_uint2(p02.y, p13.y);
    uint2 ku1 = tf2x32(ku, make_uint2(0u, 1u));            // fold_in(ku,1)
    KEYS[tid * 4 + 0] = kp.x;  KEYS[tid * 4 + 1] = kp.y;
    KEYS[tid * 4 + 2] = ku1.x; KEYS[tid * 4 + 3] = ku1.y;
    float frac = (float)tid / (float)(NSTEP - 1);
    TINV[tid] = 1.0f / __expf(logf(0.1f) * frac);          // T_INIT=1, T_FINAL=0.1
  }
  __syncthreads();

  // ---- per-row head + Metropolis sampling (threads 0..63) -------------------
  if (tid < 64) {
    const unsigned b = (unsigned)(m0 + tid);
    float pz[4];
#pragma unroll
    for (int q = 0; q < 4; ++q)
      pz[q] = PH[tid * 4 + q] + PH[(64 + tid) * 4 + q];

    float beta = log1pf(__expf(inv_temp[0]));              // softplus
    float4 w2v = *(const float4*)w2;
    float4 b1v = *(const float4*)b1;
    float  b2s = b2p[0];

    float en[ED];
    float* outE = out + (size_t)NB * ED;
    float* outL = out + (size_t)2 * NB * ED;
#pragma unroll
    for (int e = 0; e < ED; ++e) {
      const float4 wi = *(const float4*)(W1 + (HD + e) * 4);   // W1i row e
      float h0 = tanhf(tanhf(pz[0] + wi.x + b1v.x));           // double tanh
      float h1 = tanhf(tanhf(pz[1] + wi.y + b1v.y));
      float h2 = tanhf(tanhf(pz[2] + wi.z + b1v.z));
      float h3 = tanhf(tanhf(pz[3] + wi.w + b1v.w));
      float s  = h0 * w2v.x + h1 * w2v.y + h2 * w2v.z + h3 * w2v.w + b2s;
      float ee = 3.0f * tanhf(s);                              // ENERGY_SCALE
      en[e] = ee;
      outE[(size_t)b * ED + e] = ee;
      outL[(size_t)b * ED + e] = -beta * ee;
    }

    uint2 root = make_uint2(0u, 1u);
    uint2 o02 = tf2x32(root, make_uint2(0u, 2u));
    uint2 o13 = tf2x32(root, make_uint2(1u, 3u));
    uint2 kinit = make_uint2(o02.x, o13.x);
    int cur = (int)(row_bits(kinit, b) & 15u);

    for (int t = 0; t < NSTEP; ++t) {
      uint2 kp  = make_uint2(KEYS[t * 4 + 0], KEYS[t * 4 + 1]);
      uint2 ku1 = make_uint2(KEYS[t * 4 + 2], KEYS[t * 4 + 3]);
      int prop = (int)(row_bits(kp, b) & 15u);
      unsigned ub = row_bits(ku1, b);
      float u = __builtin_bit_cast(float, 0x3F800000u | (ub >> 9)) - 1.0f;
      float d = -(en[prop] - en[cur]) * TINV[t];
      float a = __expf(fminf(d, 0.0f));
      if (u < a) cur = prop;
    }
#pragma unroll
    for (int e = 0; e < ED; ++e)
      out[(size_t)b * ED + e] = (e == cur) ? 1.0f : 0.0f;
  }
}

// ---------------------------------------------------------------------------
extern "C" void kernel_launch(void* const* d_in, const int* in_sizes, int n_in,
                              void* d_out, int out_size, void* d_ws, size_t ws_size,
                              hipStream_t stream) {
  (void)in_sizes; (void)n_in; (void)out_size; (void)ws_size;
  const float* x        = (const float*)d_in[0];
  const float* We       = (const float*)d_in[1];
  const float* be       = (const float*)d_in[2];
  const float* W1       = (const float*)d_in[3];
  const float* b1       = (const float*)d_in[4];
  const float* w2       = (const float*)d_in[5];
  const float* b2       = (const float*)d_in[6];
  const float* inv_temp = (const float*)d_in[7];
  float*       out      = (float*)d_out;
  unsigned*    WeT      = (unsigned*)d_ws;          // 1 MB bf16 WeT[n][k]

  qbm_prep_weT<<<dim3((HD * VD / 2 + 255) / 256), dim3(256), 0, stream>>>(We, WeT);
  qbm_main<<<dim3(NB / 64), dim3(128), SMEM_BYTES, stream>>>(
      x, be, W1, b1, w2, b2, inv_temp, (const __bf16*)WeT, out);
}
